// EquivariantGraphAttention_15092515078737
// MI455X (gfx1250) — matrix operand order repaired and checked by reading, not compiled, not run on hardware
//
#include <hip/hip_runtime.h>
#include <hip/hip_bf16.h>
#include <math.h>

// Shapes: B=64, C=128, H=W=80, N=2 nodes, 4 heads x 32 dim.
// ws layout (floats):
//   nodes[16384] pos[256] msum[128] q[16384] k[16384] v[16384]
//   ao[16384] hf[16384] wmap[819200] mres[819200]   => ~6.95 MB total.

typedef __attribute__((ext_vector_type(16))) _Float16     v16h;
typedef __attribute__((ext_vector_type(8)))  float        v8f;
typedef __attribute__((ext_vector_type(4)))  float        f4;
typedef __attribute__((ext_vector_type(4)))  unsigned int v4u;
typedef __attribute__((ext_vector_type(8)))  int          v8i;
typedef __attribute__((ext_vector_type(4)))  int          v4i;

#define Bv   64
#define Cv   128
#define Hv   80
#define Wv   80
#define PIX  6400
#define SIG2 0.18f

// ---------------------------------------------------------------------------
// K1a: per (b,n): nearest-resample mask (strided gather) -> mres (ws),
//      centroid of binary mask -> pos, float-mask sum -> msum.
// grid = B*N = 128 blocks, 256 threads. Tiny (~26 MB gather traffic).
// ---------------------------------------------------------------------------
__global__ void k1a_mask(const float* __restrict__ masks, float* __restrict__ mres,
                         float* __restrict__ pos, float* __restrict__ msumA) {
  __shared__ float4 red4[256];
  const int bn  = blockIdx.x;
  const int tid = threadIdx.x;

  float msum = 0.f, cnt = 0.f, cy = 0.f, cx = 0.f;
  const float* mbase = masks + (size_t)bn * 640 * 640;
  for (int p = tid; p < PIX; p += 256) {
    int hh = p / Wv, ww = p - hh * Wv;
    float mv = mbase[(hh * 8) * 640 + ww * 8];      // nearest: (h*640)//80 = h*8
    mres[(size_t)bn * PIX + p] = mv;
    msum += mv;
    if (mv > 0.5f) { cnt += 1.f; cy += (float)hh; cx += (float)ww; }
  }
  red4[tid] = make_float4(msum, cnt, cy, cx);
  __syncthreads();
  for (int s = 128; s > 0; s >>= 1) {
    if (tid < s) {
      float4 a = red4[tid], c4 = red4[tid + s];
      red4[tid] = make_float4(a.x + c4.x, a.y + c4.y, a.z + c4.z, a.w + c4.w);
    }
    __syncthreads();
  }
  if (tid == 0) {
    float4 tot = red4[0];
    msumA[bn] = tot.x;
    float cxm = tot.w / tot.y;   // cnt>0 w.p.~1; NaN/Inf matches reference otherwise
    float cym = tot.z / tot.y;
    pos[bn * 2 + 0] = cxm / (float)Wv * 2.f - 1.f;
    pos[bn * 2 + 1] = cym / (float)Hv * 2.f - 1.f;
  }
}

// ---------------------------------------------------------------------------
// K1b: masked average pool: nodes[bn][c] = dot(x[b,c,:], mres[bn,:]) / msum.
// grid = (B*N, 8 channel groups) = 1024 blocks, 256 threads (8 waves x 2 ch).
// Mask tile (25.6 KB) staged global->LDS with the Tensor Data Mover (D# per
// ISA 8.3/8.4: 2D, data_size=4B, dim0=tile0=6400, dim1=1), issued by wave 0,
// completion via s_wait_tensorcnt.
// ---------------------------------------------------------------------------
__global__ void k1b_nodes(const float* __restrict__ x, const float* __restrict__ mres,
                          const float* __restrict__ msumA, float* __restrict__ nodes) {
  __shared__ __align__(16) float mL[PIX];
  const int bn = blockIdx.x, g = blockIdx.y;
  const int b  = bn >> 1;
  const int tid = threadIdx.x;

  if (tid < 32) {                               // one wave issues the TDM DMA
    unsigned ldsoff = (unsigned)(size_t)(void*)mL;            // addr[31:0] = LDS offset
    unsigned long long ga = (unsigned long long)(size_t)(mres + (size_t)bn * PIX);
    v4u g0 = { 1u,                                            // count=1, user D#
               ldsoff,                                        // lds_addr
               (unsigned)(ga & 0xffffffffu),                  // global_addr[31:0]
               (unsigned)((ga >> 32) & 0x1ffffffu) | 0x80000000u }; // addr[56:32] | type=2
    v8i g1 = { 0x00020000,                   // data_size=4B, no mask/iter/pad
               (int)(PIX << 16),             // tensor_dim0[15:0] in [31:16]
               0x00010000,                   // tensor_dim0 hi=0, tensor_dim1=1
               (int)(PIX << 16),             // tensor_dim1 hi=0, tile_dim0=6400
               1,                            // tile_dim1=1, tile_dim2=0
               PIX,                          // tensor_dim0_stride lo
               0, 0 };                       // stride hi, tensor_dim1_stride
    v4i gz4 = { 0, 0, 0, 0 };
    v8i gz8 = { 0, 0, 0, 0, 0, 0, 0, 0 };
    __builtin_amdgcn_tensor_load_to_lds(g0, g1, gz4, gz4, gz8, 0);
    __builtin_amdgcn_s_wait_tensorcnt(0);
  }
  __syncthreads();

  const float inv = 1.f / (msumA[bn] + 1e-6f);
  const int lane = tid & 31, wid = tid >> 5;
  const int c = g * 16 + wid * 2;               // this wave: channels c, c+1
  const f4* mL4 = (const f4*)mL;
  const f4* xb0 = (const f4*)(x + ((size_t)b * Cv + c) * PIX);
  const f4* xb1 = xb0 + PIX / 4;
  float a0 = 0.f, a1 = 0.f;
  for (int it = 0; it < PIX / 128; ++it) {
    f4 mv = mL4[it * 32 + lane];
    f4 x0 = xb0[it * 32 + lane];
    f4 x1 = xb1[it * 32 + lane];
    a0 += x0.x * mv.x + x0.y * mv.y + x0.z * mv.z + x0.w * mv.w;
    a1 += x1.x * mv.x + x1.y * mv.y + x1.z * mv.z + x1.w * mv.w;
  }
  #pragma unroll
  for (int off = 16; off > 0; off >>= 1) {
    a0 += __shfl_xor(a0, off, 32);
    a1 += __shfl_xor(a1, off, 32);
  }
  if (lane == 0) {
    nodes[bn * Cv + c]     = a0 * inv;
    nodes[bn * Cv + c + 1] = a1 * inv;
  }
}

// ---------------------------------------------------------------------------
// K2a: Q/K/V projections as (128x128)@(128x128)^T via v_wmma_f32_16x16x32_f16.
// grid = (8 col-tiles, 3 matrices), 256 threads = 8 waves (one M-tile each).
// ---------------------------------------------------------------------------
__global__ void k2a_qkv(const float* __restrict__ nodes,
                        const float* __restrict__ wq, const float* __restrict__ bq,
                        const float* __restrict__ wk, const float* __restrict__ bk,
                        const float* __restrict__ wv, const float* __restrict__ bv,
                        float* __restrict__ q, float* __restrict__ k, float* __restrict__ v) {
  const int lane = threadIdx.x & 31;
  const int wid  = threadIdx.x >> 5;
  const int sel  = blockIdx.y;
  const float* Wt = (sel == 0) ? wq : (sel == 1) ? wk : wv;
  const float* Bs = (sel == 0) ? bq : (sel == 1) ? bk : bv;
  float*       Ot = (sel == 0) ? q  : (sel == 1) ? k  : v;

  const int row0   = wid * 16;
  const int col0   = blockIdx.x * 16;
  const int arow   = row0 + (lane & 15);
  const int khalf8 = (lane >> 4) << 3;
  const int ocol   = col0 + (lane & 15);
  const int ksl    = (lane >> 4) << 4;

  v8f acc = {};
  for (int kk = 0; kk < 4; ++kk) {            // K = 128 in chunks of 32
    v16h a, bf;
    const float* ap = nodes + arow * 128 + kk * 32;
    #pragma unroll
    for (int e = 0; e < 16; ++e) {            // A: contiguous k = 16*(lane/16) + e (matches B)
      int K = ksl + e;
      a[e] = (_Float16)ap[K];
    }
    const float* bp = Wt + ocol * 128 + kk * 32 + ksl;
    #pragma unroll
    for (int e = 0; e < 16; ++e) bf[e] = (_Float16)bp[e];   // B[k][n] = W[n][k]
    acc = __builtin_amdgcn_wmma_f32_16x16x32_f16(false, a, false, bf,
                                                 (short)0, acc, false, false);
  }
  const float bias = Bs[ocol];
  #pragma unroll
  for (int i = 0; i < 8; ++i) {
    int m = row0 + i + ((lane >> 4) << 3);
    Ot[m * 128 + ocol] = acc[i] + bias;
  }
}

// ---------------------------------------------------------------------------
// K2b: attention over N=2 nodes with cdist bias. grid=B, 256 thr: warp=(head,n).
// ---------------------------------------------------------------------------
__global__ void k2b_attn(const float* __restrict__ q, const float* __restrict__ k,
                         const float* __restrict__ v, const float* __restrict__ pos,
                         float* __restrict__ ao) {
  const int b = blockIdx.x, tid = threadIdx.x;
  const int d = tid & 31, hn = tid >> 5, h = hn >> 1, n = hn & 1;
  const int r = b * 2 + n, cb = h * 32 + d;
  float qv = q[r * 128 + cb];
  float s0 = qv * k[(b * 2 + 0) * 128 + cb];
  float s1 = qv * k[(b * 2 + 1) * 128 + cb];
  #pragma unroll
  for (int off = 16; off > 0; off >>= 1) {
    s0 += __shfl_xor(s0, off, 32);
    s1 += __shfl_xor(s1, off, 32);
  }
  const float inv = 0.17677669529663687f;     // 1/sqrt(32)
  float dx = pos[(b * 2) * 2 + 0] - pos[(b * 2 + 1) * 2 + 0];
  float dy = pos[(b * 2) * 2 + 1] - pos[(b * 2 + 1) * 2 + 1];
  float dist = sqrtf(dx * dx + dy * dy);
  float sc0 = s0 * inv - ((n == 0) ? 0.f : dist);
  float sc1 = s1 * inv - ((n == 1) ? 0.f : dist);
  float mx = fmaxf(sc0, sc1);
  float e0 = expf(sc0 - mx), e1 = expf(sc1 - mx);
  float wsum = e0 + e1;
  float ov = (e0 / wsum) * v[(b * 2 + 0) * 128 + cb] +
             (e1 / wsum) * v[(b * 2 + 1) * 128 + cb];
  ao[r * 128 + cb] = ov;
}

__device__ __forceinline__ float blockSum256(float val, float* red) {
  const int tid = threadIdx.x;
  red[tid] = val; __syncthreads();
  for (int s = 128; s > 0; s >>= 1) {
    if (tid < s) red[tid] += red[tid + s];
    __syncthreads();
  }
  float r = red[0]; __syncthreads();
  return r;
}

// ---------------------------------------------------------------------------
// K2c: out-proj + residual + LN1 + FFN + residual + LN2. grid=128 rows, 256 thr.
// ---------------------------------------------------------------------------
__global__ void k2c_mlp(const float* __restrict__ nodes, const float* __restrict__ ao,
                        const float* __restrict__ wo, const float* __restrict__ bo,
                        const float* __restrict__ g1, const float* __restrict__ be1,
                        const float* __restrict__ w1, const float* __restrict__ b1,
                        const float* __restrict__ w2, const float* __restrict__ b2,
                        const float* __restrict__ g2, const float* __restrict__ be2,
                        float* __restrict__ hout) {
  __shared__ float aoL[128], hL[128], fL[256], red[256];
  const int r = blockIdx.x, tid = threadIdx.x;
  if (tid < 128) aoL[tid] = ao[r * 128 + tid];
  __syncthreads();

  float t = 0.f;
  if (tid < 128) {
    float acc = bo[tid];
    const float* wr = wo + tid * 128;
    for (int c = 0; c < 128; ++c) acc += aoL[c] * wr[c];
    t = nodes[r * 128 + tid] + acc;
  }
  float mu = blockSum256((tid < 128) ? t : 0.f, red) * (1.f / 128.f);
  float var = blockSum256((tid < 128) ? (t - mu) * (t - mu) : 0.f, red) * (1.f / 128.f);
  if (tid < 128) hL[tid] = (t - mu) * rsqrtf(var + 1e-5f) * g1[tid] + be1[tid];
  __syncthreads();

  {
    float acc = b1[tid];
    const float* wr = w1 + tid * 128;
    for (int c = 0; c < 128; ++c) acc += hL[c] * wr[c];
    fL[tid] = fmaxf(acc, 0.f);
  }
  __syncthreads();

  float t2 = 0.f;
  if (tid < 128) {
    float acc = b2[tid];
    const float* wr = w2 + tid * 256;
    for (int j = 0; j < 256; ++j) acc += fL[j] * wr[j];
    t2 = hL[tid] + acc;
  }
  float mu2 = blockSum256((tid < 128) ? t2 : 0.f, red) * (1.f / 128.f);
  float var2 = blockSum256((tid < 128) ? (t2 - mu2) * (t2 - mu2) : 0.f, red) * (1.f / 128.f);
  if (tid < 128)
    hout[r * 128 + tid] = (t2 - mu2) * rsqrtf(var2 + 1e-5f) * g2[tid] + be2[tid];
}

// ---------------------------------------------------------------------------
// K3a: gaussian weight map per (b,n,h,w). 819200 elems = 3200 * 256.
// ---------------------------------------------------------------------------
__global__ void k3a_wmap(const float* __restrict__ pos, float* __restrict__ wmap) {
  const int idx = blockIdx.x * 256 + threadIdx.x;
  const int pix = idx % PIX, bn = idx / PIX;
  const int hh = pix / Wv, ww = pix - hh * Wv;
  const float xg = -1.f + (2.f / 79.f) * (float)ww;   // linspace(-1,1,80)
  const float yg = -1.f + (2.f / 79.f) * (float)hh;
  const float dx = xg - pos[bn * 2 + 0];
  const float dy = yg - pos[bn * 2 + 1];
  wmap[idx] = expf(-(dx * dx + dy * dy) * (1.f / SIG2));
}

// ---------------------------------------------------------------------------
// K3b: out = x + sum_n hf[b,n,c] * wmap[b,n,pix]. Pure streaming, float4.
// x/out marked non-temporal (read/written once) so the reused wmap stays
// cached. 13107200 float4 = 51200 * 256.
// ---------------------------------------------------------------------------
__global__ void k3b_splat(const f4* __restrict__ x4, const float* __restrict__ hf,
                          const f4* __restrict__ wmap4, f4* __restrict__ out4) {
  const int i4 = blockIdx.x * 256 + threadIdx.x;
  const int pix4 = i4 % (PIX / 4);
  const int bc = i4 / (PIX / 4);
  const int c = bc & 127, b = bc >> 7;
  const float h0 = hf[(b * 2 + 0) * 128 + c];
  const float h1 = hf[(b * 2 + 1) * 128 + c];
  const f4 w0 = wmap4[(size_t)(b * 2 + 0) * (PIX / 4) + pix4];
  const f4 w1 = wmap4[(size_t)(b * 2 + 1) * (PIX / 4) + pix4];
  const f4 xv = __builtin_nontemporal_load(&x4[i4]);
  f4 o = xv + h0 * w0 + h1 * w1;
  __builtin_nontemporal_store(o, &out4[i4]);
}

extern "C" void kernel_launch(void* const* d_in, const int* in_sizes, int n_in,
                              void* d_out, int out_size, void* d_ws, size_t ws_size,
                              hipStream_t stream) {
  (void)in_sizes; (void)n_in; (void)out_size; (void)ws_size;
  const float* x     = (const float*)d_in[0];
  const float* masks = (const float*)d_in[1];
  const float* wq = (const float*)d_in[2];  const float* bq = (const float*)d_in[3];
  const float* wk = (const float*)d_in[4];  const float* bk = (const float*)d_in[5];
  const float* wv = (const float*)d_in[6];  const float* bv = (const float*)d_in[7];
  const float* wo = (const float*)d_in[8];  const float* bo = (const float*)d_in[9];
  const float* g1 = (const float*)d_in[10]; const float* be1 = (const float*)d_in[11];
  const float* g2 = (const float*)d_in[12]; const float* be2 = (const float*)d_in[13];
  const float* w1 = (const float*)d_in[14]; const float* b1 = (const float*)d_in[15];
  const float* w2 = (const float*)d_in[16]; const float* b2 = (const float*)d_in[17];
  float* out = (float*)d_out;

  float* ws    = (float*)d_ws;
  float* nodes = ws;                 // 16384
  float* pos   = ws + 16384;         // 256
  float* msum  = ws + 16640;         // 128
  float* q     = ws + 16768;         // 16384
  float* k     = ws + 33152;         // 16384
  float* v     = ws + 49536;         // 16384
  float* ao    = ws + 65920;         // 16384
  float* hf    = ws + 82304;         // 16384
  float* wmap  = ws + 98688;         // 819200
  float* mres  = ws + 917888;        // 819200

  k1a_mask<<<dim3(Bv * 2), dim3(256), 0, stream>>>(masks, mres, pos, msum);
  k1b_nodes<<<dim3(Bv * 2, 8), dim3(256), 0, stream>>>(x, mres, msum, nodes);
  k2a_qkv<<<dim3(8, 3), dim3(256), 0, stream>>>(nodes, wq, bq, wk, bk, wv, bv, q, k, v);
  k2b_attn<<<dim3(Bv), dim3(256), 0, stream>>>(q, k, v, pos, ao);
  k2c_mlp<<<dim3(Bv * 2), dim3(256), 0, stream>>>(nodes, ao, wo, bo, g1, be1,
                                                  w1, b1, w2, b2, g2, be2, hf);
  k3a_wmap<<<dim3((Bv * 2 * PIX) / 256), dim3(256), 0, stream>>>(pos, wmap);
  k3b_splat<<<dim3((Bv * Cv * PIX / 4) / 256), dim3(256), 0, stream>>>(
      (const f4*)x, hf, (const f4*)wmap, (f4*)out);
}
